// GRUModel_32461362823228
// MI455X (gfx1250) — compile-verified, matrix operand order checked by reading, not run on hardware
//
#include <hip/hip_runtime.h>
#include <hip/hip_bf16.h>

typedef __attribute__((ext_vector_type(16))) _Float16 v16h;
typedef __attribute__((ext_vector_type(8)))  float    v8f;

namespace {
constexpr int kB = 4096;
constexpr int kT = 512;
constexpr int kH = 50;

__device__ __forceinline__ v8f wmma16(v16h a, v16h b, v8f c) {
  // D = A(16x32 f16) * B(32x16 f16) + C(16x16 f32)
  return __builtin_amdgcn_wmma_f32_16x16x32_f16(false, a, false, b, (short)0, c,
                                                false, false);
}

// Single-instruction transcendental tanh (V_TANH_F32, gfx1250 TRANS op).
__device__ __forceinline__ float fast_tanh(float v) {
#if __has_builtin(__builtin_amdgcn_tanhf)
  return __builtin_amdgcn_tanhf(v);
#else
  float r;
  asm("v_tanh_f32 %0, %1" : "=v"(r) : "v"(v));
  return r;
#endif
}

// sigmoid(x) = 0.5 + 0.5*tanh(0.5*x)  -- no divide, 1 TRANS + 2 VALU.
__device__ __forceinline__ float fast_sigmoid(float v) {
  return fmaf(0.5f, fast_tanh(0.5f * v), 0.5f);
}

// Augmented weight element for gate block `blk` (0=r,1=z,2=n), padded row j,
// augmented K column kk.  K layout: [0..49]=h, 50=bias(=1), 51..53=x_t.
// mode 0: r/z combined (W_hh | b_ih+b_hh | W_ih)
// mode 1: n recurrent-only (W_hh | b_hh | 0)
// mode 2: n input-only     (0    | b_ih | W_ih)
__device__ __forceinline__ float wval(int blk, int j, int kk, int mode,
                                      const float* W_ih, const float* W_hh,
                                      const float* b_ih, const float* b_hh) {
  if (j >= kH) return 0.0f;
  const int g = blk * kH + j;
  if (mode != 2 && kk < kH) return W_hh[g * kH + kk];
  if (kk == kH) {
    if (mode == 0) return b_ih[g] + b_hh[g];
    if (mode == 1) return b_hh[g];
    return b_ih[g];
  }
  if (mode != 1 && kk > kH && kk <= kH + 3) return W_ih[g * 3 + (kk - kH - 1)];
  return 0.0f;
}
} // namespace

// One wave32 per 16 batch rows; state (h) lives in WMMA D-fragments in VGPRs
// for the whole 512-step scan.  grid = 4096/16 = 256 blocks of 32 threads.
__global__ __launch_bounds__(32, 1)
void gru_scan_wmma(const float* __restrict__ x,
                   const float* __restrict__ W_ih, const float* __restrict__ W_hh,
                   const float* __restrict__ b_ih, const float* __restrict__ b_hh,
                   const float* __restrict__ W_fc, const float* __restrict__ b_fc,
                   float* __restrict__ out) {
  const int lane = threadIdx.x & 31;
  const int m    = lane & 15;   // batch column within tile
  const int hi   = lane >> 4;   // lane half (selects K / row subset)
  const int b0   = blockIdx.x * 16;

  // ---- Build weight A-fragments once (ISA 16-bit A layout:
  //      lane holds row M=lane%16; element e -> K = base + e + 8*((e>>3)+hi))
  v16h aRZ[8][2];  // tiles 0-3: r rows 0..63(pad), tiles 4-7: z ; 2 K-chunks
  v16h aNh[4][2];  // n recurrent part
  v16h aNg[4];     // n input part (chunk1 only: K>=32 carries bias+x cols)
#pragma unroll
  for (int t = 0; t < 8; ++t) {
    const int blk = t >> 2;
    const int j = ((t & 3) << 4) + m;
#pragma unroll
    for (int c = 0; c < 2; ++c)
#pragma unroll
      for (int e = 0; e < 16; ++e) {
        const int kk = c * 32 + e + 8 * ((e >> 3) + hi);
        aRZ[t][c][e] = (_Float16)wval(blk, j, kk, 0, W_ih, W_hh, b_ih, b_hh);
      }
  }
#pragma unroll
  for (int t = 0; t < 4; ++t) {
    const int j = (t << 4) + m;
#pragma unroll
    for (int c = 0; c < 2; ++c)
#pragma unroll
      for (int e = 0; e < 16; ++e) {
        const int kk = c * 32 + e + 8 * ((e >> 3) + hi);
        aNh[t][c][e] = (_Float16)wval(2, j, kk, 1, W_ih, W_hh, b_ih, b_hh);
      }
#pragma unroll
    for (int e = 0; e < 16; ++e) {
      const int kk = 32 + e + 8 * ((e >> 3) + hi);
      aNg[t][e] = (_Float16)wval(2, j, kk, 2, W_ih, W_hh, b_ih, b_hh);
    }
  }

  // h^T in D-fragment layout: hD[tt][q] = h[row 16*tt + q + 8*hi][col b0+m]
  v8f hD[4];
#pragma unroll
  for (int tt = 0; tt < 4; ++tt)
#pragma unroll
    for (int q = 0; q < 8; ++q) hD[tt][q] = 0.0f;

  const float* xrow = x + (size_t)(b0 + m) * kT * 3;
  const v8f zero8 = {0.f, 0.f, 0.f, 0.f, 0.f, 0.f, 0.f, 0.f};

  for (int t = 0; t < kT; ++t) {
    // D-fragment -> B-fragment handoff is a pure per-lane f32->f16 convert:
    // B chunk0 elem e = hD[0][e] (e<8) / hD[1][e-8] ; chunk1 = hD[2]/hD[3].
    v16h bh0, bh1;
#pragma unroll
    for (int e = 0; e < 8; ++e) {
      bh0[e]     = (_Float16)hD[0][e];
      bh0[e + 8] = (_Float16)hD[1][e];
      bh1[e]     = (_Float16)hD[2][e];
      bh1[e + 8] = (_Float16)hD[3][e];
    }
    // Inject augmented columns (K=50..53 live only in lane-half 0):
    if (hi == 0) {
      bh1[10] = (_Float16)1.0f;                 // K=50 bias column
      bh1[11] = (_Float16)xrow[t * 3 + 0];      // K=51..53 = x_t
      bh1[12] = (_Float16)xrow[t * 3 + 1];
      bh1[13] = (_Float16)xrow[t * 3 + 2];
    }

#pragma unroll
    for (int tt = 0; tt < 4; ++tt) {
      v8f accR = wmma16(aRZ[tt][0], bh0, zero8);
      accR     = wmma16(aRZ[tt][1], bh1, accR);
      v8f accZ = wmma16(aRZ[4 + tt][0], bh0, zero8);
      accZ     = wmma16(aRZ[4 + tt][1], bh1, accZ);
      v8f giN  = wmma16(aNg[tt], bh1, zero8);
      v8f hwN  = wmma16(aNh[tt][0], bh0, zero8);
      hwN      = wmma16(aNh[tt][1], bh1, hwN);
#pragma unroll
      for (int q = 0; q < 8; ++q) {
        const float r = fast_sigmoid(accR[q]);
        const float z = fast_sigmoid(accZ[q]);
        const float n = fast_tanh(fmaf(r, hwN[q], giN[q]));
        // (1-z)*n + z*h  ==  n + z*(h - n)
        hD[tt][q] = fmaf(z, hD[tt][q] - n, n);
      }
    }
  }

  // out[b] = h_last @ W_fc^T + b_fc ; each lane holds 32 hidden rows of col m,
  // the paired lane (xor 16) holds the other rows.
  float acc = 0.0f;
#pragma unroll
  for (int tt = 0; tt < 4; ++tt)
#pragma unroll
    for (int q = 0; q < 8; ++q) {
      const int j = tt * 16 + q + 8 * hi;
      const float w = (j < kH) ? W_fc[j] : 0.0f;
      acc += w * hD[tt][q];
    }
  acc += __shfl_xor(acc, 16, 32);
  if (hi == 0) out[b0 + m] = acc + b_fc[0];
}

extern "C" void kernel_launch(void* const* d_in, const int* in_sizes, int n_in,
                              void* d_out, int out_size, void* d_ws, size_t ws_size,
                              hipStream_t stream) {
  const float* x    = (const float*)d_in[0];
  const float* W_ih = (const float*)d_in[1];
  const float* W_hh = (const float*)d_in[2];
  const float* b_ih = (const float*)d_in[3];
  const float* b_hh = (const float*)d_in[4];
  const float* W_fc = (const float*)d_in[5];
  const float* b_fc = (const float*)d_in[6];
  float* out = (float*)d_out;
  (void)in_sizes; (void)n_in; (void)out_size; (void)d_ws; (void)ws_size;
  hipLaunchKernelGGL(gru_scan_wmma, dim3(kB / 16), dim3(32), 0, stream,
                     x, W_ih, W_hh, b_ih, b_hh, W_fc, b_fc, out);
}
